// SAModule_33114197852810
// MI455X (gfx1250) — matrix operand order repaired and checked by reading, not compiled, not run on hardware
//
#include <hip/hip_runtime.h>
#include <stdint.h>

typedef __attribute__((ext_vector_type(16))) _Float16 v16h;
typedef __attribute__((ext_vector_type(8)))  float    v8f;

#define NB    8
#define NP    4096
#define MPER  1024
#define MTOT  (NB*MPER)
#define CIN   64
#define KNN   64
#define R2    0.04f

// workspace layout (bytes)
#define OFF_SEL 0
#define OFF_NBR (MTOT*4)
#define OFF_W1T (OFF_NBR + MTOT*KNN*4)      // 64 x 96 f16 (n-major, k padded 67->96)
#define OFF_W2T (OFF_W1T + 64*96*2)         // 64 x 64 f16
#define OFF_W3T (OFF_W2T + 64*64*2)         // 128 x 64 f16

// ---------------------------------------------------------------------------
// Kernel 0: convert + transpose weights to f16 (n-major so a B fragment is
// one contiguous 32B v16h load per lane).
// ---------------------------------------------------------------------------
__global__ void wconv_kernel(const float* __restrict__ W1,
                             const float* __restrict__ W2,
                             const float* __restrict__ W3,
                             _Float16* __restrict__ w1t,
                             _Float16* __restrict__ w2t,
                             _Float16* __restrict__ w3t) {
  int t = blockIdx.x * blockDim.x + threadIdx.x;
  if (t < 64*96) {                       // w1t[n][k], k in [0,96), pad zeros
    int n = t / 96, k = t % 96;
    w1t[t] = (k < 67) ? (_Float16)W1[k*64 + n] : (_Float16)0.0f;
  }
  t -= 64*96;
  if (t >= 0 && t < 64*64) {             // w2t[n][k]
    int n = t / 64, k = t % 64;
    w2t[t] = (_Float16)W2[k*64 + n];
  }
  t -= 64*64;
  if (t >= 0 && t < 128*64) {            // w3t[n][k]
    int n = t / 64, k = t % 64;
    w3t[t] = (_Float16)W3[k*128 + n];
  }
}

// ---------------------------------------------------------------------------
// Kernel 1: farthest point sampling. One 1024-thread WG per cloud; positions
// staged in LDS; per-iteration argmax via wave32 shuffles + 32-entry LDS.
// Key packs (f32 bits of dist, NP-1-idx) so a single u64 max gives
// "max dist, tie -> lowest index" (matches jnp.argmax).
// ---------------------------------------------------------------------------
__global__ __launch_bounds__(1024)
void fps_kernel(const float* __restrict__ pos, int* __restrict__ sel) {
  __shared__ float px[NP], py[NP], pz[NP];
  __shared__ unsigned long long warr[32];
  const int b = blockIdx.x;
  const int t = threadIdx.x;
  const float* p = pos + (size_t)b * NP * 3;
  for (int i = t; i < NP; i += 1024) {
    px[i] = p[i*3+0]; py[i] = p[i*3+1]; pz[i] = p[i*3+2];
  }
  __syncthreads();
  const float x0 = px[0], y0 = py[0], z0 = pz[0];
  float ptx[4], pty[4], ptz[4], mind[4];
  int   myidx[4];
  for (int i = 0; i < 4; ++i) {
    int idx = t + i*1024;
    myidx[i] = idx;
    ptx[i] = px[idx]; pty[i] = py[idx]; ptz[i] = pz[idx];
    float dx = ptx[i]-x0, dy = pty[i]-y0, dz = ptz[i]-z0;
    mind[i] = dx*dx + dy*dy + dz*dz;
  }
  if (t == 0) sel[b*MPER + 0] = 0;
  const int lane = t & 31, w = t >> 5;
  for (int it = 1; it < MPER; ++it) {
    unsigned long long k = 0ull;
    for (int i = 0; i < 4; ++i) {
      unsigned long long kk =
        ((unsigned long long)__float_as_uint(mind[i]) << 32) |
        (unsigned long long)(unsigned)(NP-1 - myidx[i]);
      k = (kk > k) ? kk : k;
    }
    for (int s = 16; s > 0; s >>= 1) {
      unsigned long long o = __shfl_xor(k, s, 32);
      k = (o > k) ? o : k;
    }
    if (lane == 0) warr[w] = k;
    __syncthreads();
    unsigned long long g = warr[0];
    for (int i = 1; i < 32; ++i) { unsigned long long o = warr[i]; g = (o > g) ? o : g; }
    const int nxt = NP-1 - (int)(g & 0xFFFFFFFFull);
    if (t == 0) sel[b*MPER + it] = nxt;
    const float nx = px[nxt], ny = py[nxt], nz = pz[nxt];
    for (int i = 0; i < 4; ++i) {
      float dx = ptx[i]-nx, dy = pty[i]-ny, dz = ptz[i]-nz;
      float d2 = dx*dx + dy*dy + dz*dz;
      mind[i] = (d2 < mind[i]) ? d2 : mind[i];
    }
    __syncthreads();   // warr consumed before next round's write
  }
}

// ---------------------------------------------------------------------------
// Kernel 2: ball query (K nearest within radius). One 256-thread WG per
// centroid; each thread keeps 16 packed (d2,idx) keys in registers; 64
// rounds of argmin with register invalidation. Also emits pos_s and batch.
// ---------------------------------------------------------------------------
__global__ __launch_bounds__(256)
void ballq_kernel(const float* __restrict__ pos, const int* __restrict__ sel,
                  int* __restrict__ nbr, float* __restrict__ pos_out,
                  float* __restrict__ batch_out) {
  __shared__ unsigned long long warr[8];
  const int c = blockIdx.x;
  const int b = c >> 10;
  const int t = threadIdx.x;
  const int sl = sel[c];
  const int sg = b*NP + sl;
  const float qx = pos[sg*3+0], qy = pos[sg*3+1], qz = pos[sg*3+2];
  if (t == 0) {
    pos_out[c*3+0] = qx; pos_out[c*3+1] = qy; pos_out[c*3+2] = qz;
    batch_out[c] = (float)b;
  }
  const float* p = pos + (size_t)b * NP * 3;
  unsigned long long ks[16];
  for (int i = 0; i < 16; ++i) {
    int idx = t + i*256;
    float dx = p[idx*3+0]-qx, dy = p[idx*3+1]-qy, dz = p[idx*3+2]-qz;
    float d2 = dx*dx + dy*dy + dz*dz;
    ks[i] = (d2 <= R2)
      ? ((unsigned long long)__float_as_uint(d2) << 32) | (unsigned long long)(unsigned)idx
      : ~0ull;
  }
  const int lane = t & 31, w = t >> 5;
  for (int r = 0; r < KNN; ++r) {
    unsigned long long k = ks[0];
    for (int i = 1; i < 16; ++i) k = (ks[i] < k) ? ks[i] : k;
    for (int s = 16; s > 0; s >>= 1) {
      unsigned long long o = __shfl_xor(k, s, 32);
      k = (o < k) ? o : k;
    }
    if (lane == 0) warr[w] = k;
    __syncthreads();
    unsigned long long g = warr[0];
    for (int i = 1; i < 8; ++i) { unsigned long long o = warr[i]; g = (o < g) ? o : g; }
    if (t == 0) nbr[c*KNN + r] = (g == ~0ull) ? -1 : (int)(g & 0xFFFFFFFFull) + b*NP;
    for (int i = 0; i < 16; ++i) if (ks[i] == g) ks[i] = ~0ull;  // owner drops winner
    __syncthreads();
  }
}

// ---------------------------------------------------------------------------
// Kernel 3: per-centroid MLP (f16 WMMA, f32 accum) + masked column max.
// 256 threads = 8 waves; tiles of v_wmma_f32_16x16x32_f16.
// A fragments: contiguous K (k = 16*hl + s), same order as B, loaded as 32-bit LDS reads.
// B fragments: single contiguous v16h load from pre-transposed f16 weights.
// ---------------------------------------------------------------------------
__global__ __launch_bounds__(256)
void mlp_kernel(const float* __restrict__ x, const float* __restrict__ pos,
                const int* __restrict__ nbr,
                const _Float16* __restrict__ w1t, const _Float16* __restrict__ w2t,
                const _Float16* __restrict__ w3t,
                const float* __restrict__ b1, const float* __restrict__ b2,
                const float* __restrict__ b3,
                const float* __restrict__ pos_s, float* __restrict__ out) {
  __shared__ _Float16 Xs[64*96];    // layer1 A (stride 96); reused as H2 (stride 64)
  __shared__ _Float16 Hs[64*64];    // H1
  __shared__ float    H3s[64*128];  // layer3 output (pre-mask)
  __shared__ int      vld[64];
  const int c = blockIdx.x;
  const int t = threadIdx.x;
  const float qx = pos_s[c*3+0], qy = pos_s[c*3+1], qz = pos_s[c*3+2];

  // ---- gather: row r = concat(x[j], pos[j]-q, zero-pad) in f16
  {
    const int r = t >> 2, sub = t & 3;
    const int j = nbr[c*KNN + r];
    if (sub == 0) vld[r] = (j >= 0);
    for (int kk = sub*24; kk < sub*24 + 24; ++kk) {
      float v = 0.f;
      if (j >= 0) {
        if (kk < CIN)       v = x[(size_t)j*CIN + kk];
        else if (kk == 64)  v = pos[j*3+0] - qx;
        else if (kk == 65)  v = pos[j*3+1] - qy;
        else if (kk == 66)  v = pos[j*3+2] - qz;
      }
      Xs[r*96 + kk] = (_Float16)v;
    }
  }
  __syncthreads();

  const int lane = t & 31, w = t >> 5;
  const int hl = lane >> 4, ln = lane & 15;   // K-half select, row/col within tile

  // ---- layer 1: X[64x96] @ W1t -> H1[64x64], relu(+b1)
  for (int tt = 0; tt < 2; ++tt) {
    const int tile = w + tt*8, mi = tile >> 2, ni = tile & 3;
    v8f acc = {};
    for (int kc = 0; kc < 3; ++kc) {
      union { v16h v; unsigned int u[8]; } A;
      const _Float16* Ap = Xs + (mi*16 + ln)*96 + kc*32 + hl*16;
      for (int j = 0; j < 8; ++j) {
        const int k2 = 2*j;        // contiguous K (k = 16*hl + s), matches B
        A.u[j] = *(const unsigned int*)(Ap + k2);
      }
      v16h Bv = *(const v16h*)(w1t + (ni*16 + ln)*96 + kc*32 + hl*16);
      acc = __builtin_amdgcn_wmma_f32_16x16x32_f16(false, A.v, false, Bv,
                                                   (short)0, acc, false, false);
    }
    const int n = ni*16 + ln;
    const float bias = b1[n];
    for (int r = 0; r < 8; ++r) {
      float v = acc[r] + bias; v = v > 0.f ? v : 0.f;
      Hs[(mi*16 + hl*8 + r)*64 + n] = (_Float16)v;
    }
  }
  __syncthreads();

  // ---- layer 2: H1 @ W2t -> H2 (into Xs region, stride 64), relu(+b2)
  _Float16* H2s = Xs;
  for (int tt = 0; tt < 2; ++tt) {
    const int tile = w + tt*8, mi = tile >> 2, ni = tile & 3;
    v8f acc = {};
    for (int kc = 0; kc < 2; ++kc) {
      union { v16h v; unsigned int u[8]; } A;
      const _Float16* Ap = Hs + (mi*16 + ln)*64 + kc*32 + hl*16;
      for (int j = 0; j < 8; ++j) {
        const int k2 = 2*j;
        A.u[j] = *(const unsigned int*)(Ap + k2);
      }
      v16h Bv = *(const v16h*)(w2t + (ni*16 + ln)*64 + kc*32 + hl*16);
      acc = __builtin_amdgcn_wmma_f32_16x16x32_f16(false, A.v, false, Bv,
                                                   (short)0, acc, false, false);
    }
    const int n = ni*16 + ln;
    const float bias = b2[n];
    for (int r = 0; r < 8; ++r) {
      float v = acc[r] + bias; v = v > 0.f ? v : 0.f;
      H2s[(mi*16 + hl*8 + r)*64 + n] = (_Float16)v;
    }
  }
  __syncthreads();

  // ---- layer 3: H2 @ W3t[128x64] -> H3[64x128] (+b3, f32)
  for (int tt = 0; tt < 4; ++tt) {
    const int tile = w*4 + tt, mi = tile >> 3, ni = tile & 7;
    v8f acc = {};
    for (int kc = 0; kc < 2; ++kc) {
      union { v16h v; unsigned int u[8]; } A;
      const _Float16* Ap = H2s + (mi*16 + ln)*64 + kc*32 + hl*16;
      for (int j = 0; j < 8; ++j) {
        const int k2 = 2*j;
        A.u[j] = *(const unsigned int*)(Ap + k2);
      }
      v16h Bv = *(const v16h*)(w3t + (ni*16 + ln)*64 + kc*32 + hl*16);
      acc = __builtin_amdgcn_wmma_f32_16x16x32_f16(false, A.v, false, Bv,
                                                   (short)0, acc, false, false);
    }
    const int n = ni*16 + ln;
    const float bias = b3[n];
    for (int r = 0; r < 8; ++r)
      H3s[(mi*16 + hl*8 + r)*128 + n] = acc[r] + bias;
  }
  __syncthreads();

  // ---- masked max over the 64 neighbor rows (order-invariant)
  if (t < 128) {
    float m = -1e30f;
    for (int r = 0; r < 64; ++r)
      if (vld[r]) { float v = H3s[r*128 + t]; m = v > m ? v : m; }
    out[(size_t)c*128 + t] = m;
  }
}

// ---------------------------------------------------------------------------
extern "C" void kernel_launch(void* const* d_in, const int* in_sizes, int n_in,
                              void* d_out, int out_size, void* d_ws, size_t ws_size,
                              hipStream_t stream) {
  const float* x   = (const float*)d_in[0];
  const float* pos = (const float*)d_in[1];
  // d_in[2] = batch (derived analytically), d_in[9] = num_graphs (fixed at 8)
  const float* W1 = (const float*)d_in[3];
  const float* b1 = (const float*)d_in[4];
  const float* W2 = (const float*)d_in[5];
  const float* b2 = (const float*)d_in[6];
  const float* W3 = (const float*)d_in[7];
  const float* b3 = (const float*)d_in[8];

  char* ws = (char*)d_ws;
  int*      sel = (int*)(ws + OFF_SEL);
  int*      nbr = (int*)(ws + OFF_NBR);
  _Float16* w1t = (_Float16*)(ws + OFF_W1T);
  _Float16* w2t = (_Float16*)(ws + OFF_W2T);
  _Float16* w3t = (_Float16*)(ws + OFF_W3T);

  float* out_x   = (float*)d_out;                 // [MTOT,128]
  float* out_pos = out_x + (size_t)MTOT*128;      // [MTOT,3]
  float* out_b   = out_pos + (size_t)MTOT*3;      // [MTOT]

  wconv_kernel<<<(64*96 + 64*64 + 128*64 + 255)/256, 256, 0, stream>>>(
      W1, W2, W3, w1t, w2t, w3t);
  fps_kernel<<<NB, 1024, 0, stream>>>(pos, sel);
  ballq_kernel<<<MTOT, 256, 0, stream>>>(pos, sel, nbr, out_pos, out_b);
  mlp_kernel<<<MTOT, 256, 0, stream>>>(x, pos, nbr, w1t, w2t, w3t,
                                       b1, b2, b3, out_pos, out_x);
}
